// VanillaRNN_59708635349023
// MI455X (gfx1250) — matrix operand order repaired and checked by reading, not compiled, not run on hardware
//
#include <hip/hip_runtime.h>

#define Bsz 512
#define Tsz 2048
#define Isz 33
#define Hsz 64
#define Osz 2
#define WPRE 8          // TDM prefetch depth (LDS ring slots)
#define ROWPAD 68       // 64 floats + 4-DWORD pad per LDS row (272 B)

typedef __attribute__((ext_vector_type(16))) _Float16 v16h;
typedef __attribute__((ext_vector_type(8)))  float v8f;
typedef __attribute__((ext_vector_type(4)))  float v4f;
typedef __attribute__((ext_vector_type(2)))  float v2f;
typedef __attribute__((ext_vector_type(4)))  unsigned int v4u;
typedef __attribute__((ext_vector_type(8)))  int v8i;
typedef __attribute__((ext_vector_type(4)))  int v4i;

union HF { v16h h; unsigned int u[8]; };

__device__ __forceinline__ v8f wmma16(v16h a, v16h b, v8f c) {
  return __builtin_amdgcn_wmma_f32_16x16x32_f16(false, a, false, b, (short)0, c,
                                                false, false);
}

__device__ __forceinline__ float tanh_fast(float x) {
#if __has_builtin(__builtin_amdgcn_tanhf)
  return __builtin_amdgcn_tanhf(x);
#else
  // tanh(x) = 1 - 2/(exp2(2x*log2e)+1)
  float e = __builtin_amdgcn_exp2f(x * 2.8853900817779268f);
  return 1.0f - 2.0f * __builtin_amdgcn_rcpf(e + 1.0f);
#endif
}

__device__ __forceinline__ void wait_ds0() {
#if __has_builtin(__builtin_amdgcn_s_wait_dscnt)
  __builtin_amdgcn_s_wait_dscnt(0);
#else
  asm volatile("s_wait_dscnt 0" ::: "memory");
#endif
}

#define WAIT_TENSOR(n) __builtin_amdgcn_s_wait_tensorcnt(n)

// ---------------------------------------------------------------------------
// Phase 1: xw[b,t,:] = x[b,t,:] @ W_ih^T + b_ih + b_hh   (written into hidden
// region of d_out; phase 2 consumes it via TDM and overwrites with h states)
// ---------------------------------------------------------------------------
__global__ __launch_bounds__(256) void rnn_inproj(
    const float* __restrict__ x, const float* __restrict__ W_ih,
    const float* __restrict__ b_ih, const float* __restrict__ b_hh,
    float* __restrict__ xw) {
  const int lane = threadIdx.x & 31;
  const int lo = lane & 15;
  const int hf = lane >> 4;
  const int wave = blockIdx.x * 8 + (threadIdx.x >> 5);

  // B fragments of W_ih^T: B[i, h] = W_ih[h, i]; K=i (padded 33->64), N=h
  HF Bf[2][4];
#pragma unroll
  for (int k = 0; k < 2; ++k)
#pragma unroll
    for (int j = 0; j < 4; ++j) {
      const float* wcol = W_ih + (16 * j + lo) * Isz;
#pragma unroll
      for (int v = 0; v < 8; ++v) {
        int i0 = 32 * k + 16 * hf + 2 * v;
        float f0 = (i0 < Isz) ? wcol[i0] : 0.f;
        float f1 = (i0 + 1 < Isz) ? wcol[i0 + 1] : 0.f;
        Bf[k][j].h[2 * v] = (_Float16)f0;
        Bf[k][j].h[2 * v + 1] = (_Float16)f1;
      }
    }
  float bias[4];
#pragma unroll
  for (int j = 0; j < 4; ++j) bias[j] = b_ih[16 * j + lo] + b_hh[16 * j + lo];

  for (int tile = wave; tile < (Bsz * Tsz / 16); tile += 8192) {
    const float* xr = x + (long)(16 * tile + lo) * Isz;
    HF Af[2];  // A = x tile rows, K=i
#pragma unroll
    for (int k = 0; k < 2; ++k)
#pragma unroll
      for (int e = 0; e < 8; ++e) {
        int c0 = 32 * k + 16 * hf + e;
        int c1 = c0 + 8;
        Af[k].h[e] = (_Float16)((c0 < Isz) ? xr[c0] : 0.f);
        Af[k].h[e + 8] = (_Float16)((c1 < Isz) ? xr[c1] : 0.f);
      }
    v8f acc[4];
#pragma unroll
    for (int j = 0; j < 4; ++j) {
#pragma unroll
      for (int e = 0; e < 8; ++e) acc[j][e] = bias[j];
      acc[j] = wmma16(Af[0].h, Bf[0][j].h, acc[j]);
      acc[j] = wmma16(Af[1].h, Bf[1][j].h, acc[j]);
    }
    float* orow = xw + (long)(16 * tile) * Hsz;
#pragma unroll
    for (int r = 0; r < 8; ++r)
#pragma unroll
      for (int j = 0; j < 4; ++j)
        orow[(r + 8 * hf) * Hsz + 16 * j + lo] = acc[j][r];
  }
}

// ---------------------------------------------------------------------------
// Phase 2: sequential scan. One wave per 16-batch tile (32 waves total).
// h^T update: D[m] = sum_k WMMA(A=W_hh[m][k], B=h^T[k]) + xw (C operand).
// xw staged global->LDS via TDM ring buffer, depth WPRE.
// ---------------------------------------------------------------------------
__device__ __forceinline__ void tdm_issue(unsigned lds_addr,
                                          unsigned long long ga, v8i g1) {
  v4u g0;
  g0[0] = 1u;                                             // count=1
  g0[1] = lds_addr;                                       // lds byte addr
  g0[2] = (unsigned)ga;                                   // global addr lo
  g0[3] = (unsigned)((ga >> 32) & 0x01FFFFFFu) | 0x80000000u;  // hi | type=2
  v4i gz = {0, 0, 0, 0};
  v8i gz8 = {0, 0, 0, 0, 0, 0, 0, 0};
  __builtin_amdgcn_tensor_load_to_lds(g0, g1, gz, gz, gz8, 0);
}

__global__ __launch_bounds__(32) void rnn_scan(const float* __restrict__ h0,
                                               const float* __restrict__ W_hh,
                                               float* __restrict__ hid) {
  __shared__ float smem[WPRE * 16 * ROWPAD];  // 34,816 B
  const int lane = threadIdx.x & 31;
  const int lo = lane & 15;
  const int hf = lane >> 4;
  const int btile = blockIdx.x * 16;

  // A fragments of W_hh (resident for whole scan): lane = row M=16m+lo
  HF Af[4][2];
#pragma unroll
  for (int m = 0; m < 4; ++m)
#pragma unroll
    for (int k = 0; k < 2; ++k) {
      const float* wr = W_hh + (16 * m + lo) * Hsz + 32 * k + 16 * hf;
#pragma unroll
      for (int e = 0; e < 8; ++e) {
        Af[m][k].h[e] = (_Float16)wr[e];
        Af[m][k].h[e + 8] = (_Float16)wr[e + 8];
      }
    }
  // B fragments from h0 (h^T: K = h-dim, N = batch lane)
  HF Bf[2];
#pragma unroll
  for (int k = 0; k < 2; ++k)
#pragma unroll
    for (int v = 0; v < 8; ++v) {
      int d = 32 * k + 16 * hf + 2 * v;
      Bf[k].h[2 * v] = (_Float16)h0[(btile + lo) * Hsz + d];
      Bf[k].h[2 * v + 1] = (_Float16)h0[(btile + lo) * Hsz + d + 1];
    }

  // TDM descriptor: 2-D tile, 16 rows (batch) x 64 f32, row stride T*H,
  // LDS pad 4 DW every 64 DW -> 272B LDS rows (bank-conflict-free b128 reads)
  const unsigned lds_base = (unsigned)(unsigned long long)(const void*)smem;
  const unsigned long long gbase =
      (unsigned long long)hid + ((unsigned long long)btile * Tsz) * (Hsz * 4ull);
  v8i g1;
  g1[0] = 0x07520000;      // data_size=4B, pad_en, pad_interval=64DW, pad=4DW
  g1[1] = (Hsz << 16);     // tensor_dim0 = 64
  g1[2] = (16 << 16);      // tensor_dim1 = 16
  g1[3] = (Hsz << 16);     // tile_dim0 = 64
  g1[4] = 16;              // tile_dim1 = 16
  g1[5] = Tsz * Hsz;       // tensor_dim0_stride (elements)
  g1[6] = 0;
  g1[7] = 0;

#pragma unroll
  for (int p = 0; p < WPRE; ++p)
    tdm_issue(lds_base + (unsigned)p * 16 * 272,
              gbase + (unsigned long long)p * (Hsz * 4), g1);
  WAIT_TENSOR(WPRE - 1);  // slot 0 ready

  float* hrowbase = hid + ((long)(btile + lo)) * Tsz * Hsz + 8 * hf;

  for (int t = 0; t < Tsz; ++t) {
    const int slot = t & (WPRE - 1);
    const float* srow = &smem[slot * (16 * ROWPAD) + lo * ROWPAD + 8 * hf];
    v8f acc[4];
#pragma unroll
    for (int m = 0; m < 4; ++m) {
      v4f a0 = *(const v4f*)(srow + 16 * m);
      v4f a1 = *(const v4f*)(srow + 16 * m + 4);
      acc[m][0] = a0[0]; acc[m][1] = a0[1]; acc[m][2] = a0[2]; acc[m][3] = a0[3];
      acc[m][4] = a1[0]; acc[m][5] = a1[1]; acc[m][6] = a1[2]; acc[m][7] = a1[3];
    }
#pragma unroll
    for (int m = 0; m < 4; ++m) {
      acc[m] = wmma16(Af[m][0].h, Bf[0].h, acc[m]);
      acc[m] = wmma16(Af[m][1].h, Bf[1].h, acc[m]);
    }
    // LDS slot consumed -> refill it WPRE steps ahead
    if (t < Tsz - WPRE) {
      wait_ds0();
      tdm_issue(lds_base + (unsigned)slot * 16 * 272,
                gbase + (unsigned long long)(t + WPRE) * (Hsz * 4), g1);
    }
    // activation
#pragma unroll
    for (int m = 0; m < 4; ++m)
#pragma unroll
      for (int e = 0; e < 8; ++e) acc[m][e] = tanh_fast(acc[m][e]);
    // store hidden state (overwrites consumed xw row t)
    float* hrow = hrowbase + (long)t * Hsz;
#pragma unroll
    for (int m = 0; m < 4; ++m) {
      v4f s0 = {acc[m][0], acc[m][1], acc[m][2], acc[m][3]};
      v4f s1 = {acc[m][4], acc[m][5], acc[m][6], acc[m][7]};
      *(v4f*)(hrow + 16 * m) = s0;
      *(v4f*)(hrow + 16 * m + 4) = s1;
    }
    // D-frag -> B-frag for next step: pack f16 pairs, SWAPX16 half-wave swap
    unsigned pk[4][4];
#pragma unroll
    for (int m = 0; m < 4; ++m)
#pragma unroll
      for (int j = 0; j < 4; ++j) {
        auto p = __builtin_amdgcn_cvt_pkrtz(acc[m][2 * j], acc[m][2 * j + 1]);
        pk[m][j] = __builtin_bit_cast(unsigned, p);
      }
#pragma unroll
    for (int k = 0; k < 2; ++k)
#pragma unroll
      for (int v = 0; v < 4; ++v) {
        unsigned sl = (unsigned)__builtin_amdgcn_ds_swizzle((int)pk[2 * k][v], 0x401f);
        unsigned sh = (unsigned)__builtin_amdgcn_ds_swizzle((int)pk[2 * k + 1][v], 0x401f);
        Bf[k].u[v]     = hf ? sh : pk[2 * k][v];
        Bf[k].u[v + 4] = hf ? pk[2 * k + 1][v] : sl;
      }
    // slot t+1 must be resident before next iteration
    if (t < Tsz - WPRE) WAIT_TENSOR(WPRE - 1);
    else                WAIT_TENSOR(0);
  }
}

// ---------------------------------------------------------------------------
// Phase 3: predictions = hidden @ W_ro^T + b_ro  (memory-bound)
// ---------------------------------------------------------------------------
__global__ __launch_bounds__(256) void rnn_out(const float* __restrict__ hid,
                                               const float* __restrict__ W_ro,
                                               const float* __restrict__ b_ro,
                                               float* __restrict__ pred) {
  long row = (long)blockIdx.x * blockDim.x + threadIdx.x;
  if (row >= (long)Bsz * Tsz) return;
  const float* h = hid + row * Hsz;
  float p0 = b_ro[0], p1 = b_ro[1];
#pragma unroll
  for (int i = 0; i < Hsz; i += 4) {
    v4f hv = *(const v4f*)(h + i);
#pragma unroll
    for (int q = 0; q < 4; ++q) {
      p0 += hv[q] * W_ro[i + q];
      p1 += hv[q] * W_ro[Hsz + i + q];
    }
  }
  v2f o = {p0, p1};
  *(v2f*)(pred + row * 2) = o;
}

extern "C" void kernel_launch(void* const* d_in, const int* in_sizes, int n_in,
                              void* d_out, int out_size, void* d_ws,
                              size_t ws_size, hipStream_t stream) {
  const float* x    = (const float*)d_in[0];
  const float* h0   = (const float*)d_in[1];
  const float* W_ih = (const float*)d_in[2];
  const float* W_hh = (const float*)d_in[3];
  const float* b_ih = (const float*)d_in[4];
  const float* b_hh = (const float*)d_in[5];
  const float* W_ro = (const float*)d_in[6];
  const float* b_ro = (const float*)d_in[7];

  float* pred = (float*)d_out;
  float* hid  = pred + (size_t)Bsz * Tsz * Osz;  // hidden_states region

  rnn_inproj<<<1024, 256, 0, stream>>>(x, W_ih, b_ih, b_hh, hid);
  rnn_scan<<<Bsz / 16, 32, 0, stream>>>(h0, W_hh, hid);
  rnn_out<<<(Bsz * Tsz) / 256, 256, 0, stream>>>(hid, W_ro, b_ro, pred);
}
